// GATModel_42339787604168
// MI455X (gfx1250) — matrix operand order repaired and checked by reading, not compiled, not run on hardware
//
#include <hip/hip_runtime.h>
#include <hip/hip_bf16.h>
#include <math.h>

typedef __attribute__((ext_vector_type(16))) __bf16 v16bf;
typedef __attribute__((ext_vector_type(8)))  float  v8f;

#define EDGE_DIM 35
#define LEAKY 0.2f

__device__ __forceinline__ unsigned pack_bf16x2(float x, float y) {
  union { __bf16 b[2]; unsigned u; } p;
  p.b[0] = (__bf16)x;
  p.b[1] = (__bf16)y;
  return p.u;   // lowers to v_cvt_pk_bf16_f32
}

__device__ __forceinline__ void atomicMaxF(float* addr, float v) {
  // IEEE ordering trick: positives via signed-int max, negatives via unsigned min.
  if (v >= 0.0f) atomicMax((int*)addr, __float_as_int(v));
  else           atomicMin((unsigned int*)addr, __float_as_uint(v));
}

__global__ void fill_kernel(float* p, size_t n, float v) {
  size_t i  = (size_t)blockIdx.x * blockDim.x + threadIdx.x;
  size_t st = (size_t)gridDim.x * blockDim.x;
  for (; i < n; i += st) p[i] = v;
}

// deg[dst] += 1 ; attrsum[dst,:] += edge_attr[e,:]
__global__ void deg_attrsum_kernel(const int* __restrict__ dstA, const float* __restrict__ eattr,
                                   float* deg, float* asum, int E) {
  int e = blockIdx.x * blockDim.x + threadIdx.x;
  if (e >= E) return;
  int d = dstA[e];
  atomicAdd(&deg[d], 1.0f);
  const float* a = eattr + (size_t)e * EDGE_DIM;
  float* s = asum + (size_t)d * EDGE_DIM;
  #pragma unroll
  for (int k = 0; k < EDGE_DIM; ++k) atomicAdd(&s[k], a[k]);
}

// attrsum[n,k] /= max(deg[n],1)  -> self-loop attr, in place
__global__ void selfdiv_kernel(float* asum, const float* deg, int N) {
  int i = blockIdx.x * blockDim.x + threadIdx.x;
  if (i >= N * EDGE_DIM) return;
  asum[i] /= fmaxf(deg[i / EDGE_DIM], 1.0f);
}

// C[M,Nn] = A[M,K] @ B[K,Nn] + bias.  f32 in/out, bf16 WMMA compute.
// Block = 256 threads = 8 waves; block tile = 16(M) x 128(N); wave tile = 16x16.
// LDS holds packed bf16x2 dwords so fragments load as single ds_load_b32 each.
__global__ void __launch_bounds__(256)
gemm_bf16_wmma(const float* __restrict__ A, const float* __restrict__ B,
               const float* __restrict__ bias, float* __restrict__ C,
               int M, int Nn, int K) {
  // As32[m][q] = bf16x2 of A[m][2q],A[m][2q+1] for K-chunk of 64 (q = 0..31)
  // Bs32[q][n] = bf16x2 of B[2q][n],B[2q+1][n]                   (q = 0..31)
  __shared__ unsigned As32[16][33];    // pad 33: lane-m stride hits 16 distinct banks
  __shared__ unsigned Bs32[32][130];   // pad 130: lane halves +4 rows => +8 banks apart
  int tid  = threadIdx.x;
  int lane = tid & 31, w = tid >> 5;
  int m = lane & 15, half = lane >> 4;
  int rowBase = blockIdx.y * 16;
  int colBase = blockIdx.x * 128;
  v8f acc = {0.f, 0.f, 0.f, 0.f, 0.f, 0.f, 0.f, 0.f};
  for (int k0 = 0; k0 < K; k0 += 64) {
    // stage A: 16 rows x 32 pairs = 512 pairs, 2 per thread, float2 loads
    #pragma unroll
    for (int it = 0; it < 2; ++it) {
      int i = tid + it * 256;
      int r = i >> 5, q = i & 31;
      float2 f = *(const float2*)(A + (size_t)(rowBase + r) * K + k0 + 2 * q);
      As32[r][q] = pack_bf16x2(f.x, f.y);
    }
    // stage B: 32 pair-rows x 128 cols = 4096 pairs, 16 per thread
    #pragma unroll
    for (int it = 0; it < 16; ++it) {
      int i = tid + it * 256;
      int q = i >> 7, n = i & 127;
      const float* bp = B + (size_t)(k0 + 2 * q) * Nn + colBase + n;
      Bs32[q][n] = pack_bf16x2(bp[0], bp[Nn]);
    }
    __syncthreads();
    #pragma unroll
    for (int kk = 0; kk < 2; ++kk) {
      union { v16bf v; unsigned u[8]; } a, b;
      #pragma unroll
      for (int v = 0; v < 8; ++v) {
        // A 16x32 layout (ISA 7.12.2): pair index (v>>2)*8 + half*4 + (v&3)
        a.u[v] = As32[m][kk * 16 + (v >> 2) * 8 + half * 4 + (v & 3)];
        // B 32x16 layout: lane = N column; pair index (v>>2)*8 + half*4 + (v&3)
        b.u[v] = Bs32[kk * 16 + (v >> 2) * 8 + half * 4 + (v & 3)][w * 16 + m];
      }
      acc = __builtin_amdgcn_wmma_f32_16x16x32_bf16(false, a.v, false, b.v, (short)0, acc,
                                                    false, false);
    }
    __syncthreads();
  }
  // C/D layout: VGPR r -> M = r + 8*(lane>=16), N = lane&15
  int cn = colBase + w * 16 + m;
  int rm = rowBase + half * 8;
  float bv = bias[cn];
  #pragma unroll
  for (int r = 0; r < 8; ++r)
    C[(size_t)(rm + r) * Nn + cn] = acc[r] + bv;
}

// Attention logits: alpha[e,h] = att_h . leakyrelu(xl[src]+xr[dst]+attr@We), amax via atomics.
// One wave per edge; We (+att) staged in LDS (147KB layer1 / 74KB layer2, fits 320KB WGP LDS).
__global__ void edge_alpha_kernel(const int* __restrict__ srcA, const int* __restrict__ dstA,
                                  int E, int Etot,
                                  const float* __restrict__ eattr, const float* __restrict__ selfattr,
                                  const float* __restrict__ xl, const float* __restrict__ xr,
                                  const float* __restrict__ Weg, const float* __restrict__ att,
                                  float* __restrict__ alpha, float* __restrict__ amax,
                                  int H, int C, int logC, int d) {
  extern __shared__ float sWe[];
  float* sAtt = sWe + EDGE_DIM * d;
  for (int i = threadIdx.x; i < EDGE_DIM * d; i += blockDim.x) sWe[i] = Weg[i];
  for (int i = threadIdx.x; i < d; i += blockDim.x) sAtt[i] = att[i];
  __syncthreads();
  int lane = threadIdx.x & 31;
  int wpb = blockDim.x >> 5;
  int gw = blockIdx.x * wpb + (threadIdx.x >> 5);
  int gstride = gridDim.x * wpb;
  for (int et = gw; et < Etot; et += gstride) {
    int src, dst; const float* ap;
    if (et < E) { src = srcA[et]; dst = dstA[et]; ap = eattr + (size_t)et * EDGE_DIM; }
    else        { src = et - E;  dst = src;      ap = selfattr + (size_t)(et - E) * EDGE_DIM; }
    float ea[EDGE_DIM];
    #pragma unroll
    for (int k = 0; k < EDGE_DIM; ++k) ea[k] = ap[k];
    const float* xls = xl + (size_t)src * d;
    const float* xrs = xr + (size_t)dst * d;
    for (int h = 0; h < H; ++h) {
      float aa = 0.f;
      for (int cc = lane; cc < C; cc += 32) {
        int c = (h << logC) + cc;
        float ee = 0.f;
        #pragma unroll
        for (int k = 0; k < EDGE_DIM; ++k) ee += ea[k] * sWe[k * d + c];
        float vv = xls[c] + xrs[c] + ee;
        vv = vv > 0.f ? vv : LEAKY * vv;
        aa += vv * sAtt[c];
      }
      #pragma unroll
      for (int off = 16; off > 0; off >>= 1) aa += __shfl_xor(aa, off, 32);
      if (lane == 0) {
        alpha[(size_t)et * H + h] = aa;
        atomicMaxF(&amax[(size_t)dst * H + h], aa);
      }
    }
  }
}

// expa = exp(alpha - amax[dst]) (stored in place); den[dst,h] += expa
__global__ void edge_expsum_kernel(const int* __restrict__ dstA, int E, int Etot,
                                   float* alpha, const float* amax, float* den, int H) {
  int et = blockIdx.x * blockDim.x + threadIdx.x;
  if (et >= Etot) return;
  int dst = (et < E) ? dstA[et] : (et - E);
  for (int h = 0; h < H; ++h) {
    float a  = alpha[(size_t)et * H + h];
    float ex = __expf(a - amax[(size_t)dst * H + h]);
    alpha[(size_t)et * H + h] = ex;
    atomicAdd(&den[(size_t)dst * H + h], ex);
  }
}

// acc[dst,:] += expa[e,h] * xl[src,:]  (normalize by den afterwards, avoiding a 3rd sweep)
__global__ void edge_scatter_kernel(const int* __restrict__ srcA, const int* __restrict__ dstA,
                                    int E, int Etot,
                                    const float* __restrict__ xl, const float* __restrict__ alpha,
                                    float* __restrict__ acc, int H, int logC, int d) {
  int lane = threadIdx.x & 31;
  int wpb = blockDim.x >> 5;
  int gw = blockIdx.x * wpb + (threadIdx.x >> 5);
  int gstride = gridDim.x * wpb;
  int nq = d >> 2;
  for (int et = gw; et < Etot; et += gstride) {
    int src, dst;
    if (et < E) { src = srcA[et]; dst = dstA[et]; }
    else        { src = et - E;  dst = src; }
    const float4* xv = (const float4*)(xl + (size_t)src * d);
    float* o = acc + (size_t)dst * d;
    const float* al = alpha + (size_t)et * H;
    for (int q = lane; q < nq; q += 32) {
      float4 v = xv[q];
      int c = q << 2;
      float a = al[c >> logC];
      atomicAdd(&o[c + 0], a * v.x);
      atomicAdd(&o[c + 1], a * v.y);
      atomicAdd(&o[c + 2], a * v.z);
      atomicAdd(&o[c + 3], a * v.w);
    }
  }
}

// h = relu(acc/den + bias), in place
__global__ void finalize_kernel(float* acc, const float* den, const float* bias,
                                int N, int H, int logC, int d) {
  size_t i = (size_t)blockIdx.x * blockDim.x + threadIdx.x;
  if (i >= (size_t)N * d) return;
  int n = (int)(i / d), c = (int)(i % d);
  float v = acc[i] / den[(size_t)n * H + (c >> logC)] + bias[c];
  acc[i] = v > 0.f ? v : 0.f;
}

__global__ void colsum_kernel(const float* __restrict__ h2, float* pool, int N, int d) {
  int c = threadIdx.x;
  if (c >= d) return;
  float s = 0.f;
  for (int r = blockIdx.x; r < N; r += gridDim.x) s += h2[(size_t)r * d + c];
  atomicAdd(&pool[c], s);
}

__global__ void head_kernel(const float* __restrict__ pool, const float* __restrict__ W,
                            const float* __restrict__ b, float* out, int K, int Nout, float invN) {
  int j = threadIdx.x;
  if (j >= Nout) return;
  float s = 0.f;
  for (int k = 0; k < K; ++k) s += pool[k] * W[(size_t)k * Nout + j];
  out[j] = s * invN + b[j];
}

extern "C" void kernel_launch(void* const* d_in, const int* in_sizes, int n_in,
                              void* d_out, int out_size, void* d_ws, size_t ws_size,
                              hipStream_t stream) {
  (void)n_in; (void)out_size; (void)ws_size;
  const float* x     = (const float*)d_in[0];
  const int*   ei    = (const int*)d_in[1];
  const float* eattr = (const float*)d_in[2];
  const float* Wl1   = (const float*)d_in[3];
  const float* bl1   = (const float*)d_in[4];
  const float* Wr1   = (const float*)d_in[5];
  const float* br1   = (const float*)d_in[6];
  const float* We1   = (const float*)d_in[7];
  const float* att1  = (const float*)d_in[8];
  const float* bias1 = (const float*)d_in[9];
  const float* Wl2   = (const float*)d_in[10];
  const float* bl2   = (const float*)d_in[11];
  const float* Wr2   = (const float*)d_in[12];
  const float* br2   = (const float*)d_in[13];
  const float* We2   = (const float*)d_in[14];
  const float* att2  = (const float*)d_in[15];
  const float* bias2 = (const float*)d_in[16];
  const float* Wlin  = (const float*)d_in[17];
  const float* blin  = (const float*)d_in[18];
  float* out = (float*)d_out;

  const int IN = 1024, D1 = 1024, D2 = 512;
  const int H1 = 8, C1 = 128, LC1 = 7;
  const int H2 = 1, C2 = 512, LC2 = 9;
  int N = in_sizes[0] / IN;
  int E = in_sizes[1] / 2;
  int Etot = E + N;
  const int* srcA = ei;
  const int* dstA = ei + E;

  // Workspace carve-up (~210 MB). acc aliases the dead xr buffer each layer.
  float* ws = (float*)d_ws;
  size_t o = 0;
  float* xl1    = ws + o; o += (size_t)N * D1;
  float* xr1    = ws + o; o += (size_t)N * D1;   // also acc1 / h1
  float* xl2    = ws + o; o += (size_t)N * D2;
  float* xr2    = ws + o; o += (size_t)N * D2;   // also acc2 / h2
  float* alpha1 = ws + o; o += (size_t)Etot * H1;
  float* alpha2 = ws + o; o += (size_t)Etot * H2;
  float* amax1  = ws + o; o += (size_t)N * H1;
  float* den1   = ws + o; o += (size_t)N * H1;
  float* amax2  = ws + o; o += (size_t)N * H2;
  float* den2   = ws + o; o += (size_t)N * H2;
  float* deg    = ws + o; o += (size_t)N;
  float* asum   = ws + o; o += (size_t)N * EDGE_DIM;
  float* pool   = ws + o; o += 512;

  const int T = 256;
  const float NINF = -__builtin_huge_valf();

  // self-loop attributes: mean of incoming edge attrs
  fill_kernel<<<256, T, 0, stream>>>(deg, (size_t)N, 0.f);
  fill_kernel<<<512, T, 0, stream>>>(asum, (size_t)N * EDGE_DIM, 0.f);
  deg_attrsum_kernel<<<(E + T - 1) / T, T, 0, stream>>>(dstA, eattr, deg, asum, E);
  selfdiv_kernel<<<(N * EDGE_DIM + T - 1) / T, T, 0, stream>>>(asum, deg, N);

  // ---- layer 1 ----
  dim3 g1(D1 / 128, N / 16);
  gemm_bf16_wmma<<<g1, 256, 0, stream>>>(x, Wl1, bl1, xl1, N, D1, IN);
  gemm_bf16_wmma<<<g1, 256, 0, stream>>>(x, Wr1, br1, xr1, N, D1, IN);

  fill_kernel<<<256, T, 0, stream>>>(amax1, (size_t)N * H1, NINF);
  fill_kernel<<<256, T, 0, stream>>>(den1, (size_t)N * H1, 0.f);
  edge_alpha_kernel<<<2048, 256, (EDGE_DIM + 1) * D1 * sizeof(float), stream>>>(
      srcA, dstA, E, Etot, eattr, asum, xl1, xr1, We1, att1, alpha1, amax1, H1, C1, LC1, D1);
  fill_kernel<<<2048, T, 0, stream>>>(xr1, (size_t)N * D1, 0.f);  // acc1 (xr1 now dead)
  edge_expsum_kernel<<<(Etot + T - 1) / T, T, 0, stream>>>(dstA, E, Etot, alpha1, amax1, den1, H1);
  edge_scatter_kernel<<<2048, 256, 0, stream>>>(srcA, dstA, E, Etot, xl1, alpha1, xr1, H1, LC1, D1);
  finalize_kernel<<<(int)(((size_t)N * D1 + T - 1) / T), T, 0, stream>>>(xr1, den1, bias1, N, H1, LC1, D1);
  // h1 lives in xr1

  // ---- layer 2 ----
  dim3 g2(D2 / 128, N / 16);
  gemm_bf16_wmma<<<g2, 256, 0, stream>>>(xr1, Wl2, bl2, xl2, N, D2, D1);
  gemm_bf16_wmma<<<g2, 256, 0, stream>>>(xr1, Wr2, br2, xr2, N, D2, D1);

  fill_kernel<<<64, T, 0, stream>>>(amax2, (size_t)N * H2, NINF);
  fill_kernel<<<64, T, 0, stream>>>(den2, (size_t)N * H2, 0.f);
  edge_alpha_kernel<<<2048, 256, (EDGE_DIM + 1) * D2 * sizeof(float), stream>>>(
      srcA, dstA, E, Etot, eattr, asum, xl2, xr2, We2, att2, alpha2, amax2, H2, C2, LC2, D2);
  fill_kernel<<<2048, T, 0, stream>>>(xr2, (size_t)N * D2, 0.f);  // acc2 (xr2 now dead)
  edge_expsum_kernel<<<(Etot + T - 1) / T, T, 0, stream>>>(dstA, E, Etot, alpha2, amax2, den2, H2);
  edge_scatter_kernel<<<2048, 256, 0, stream>>>(srcA, dstA, E, Etot, xl2, alpha2, xr2, H2, LC2, D2);
  finalize_kernel<<<(int)(((size_t)N * D2 + T - 1) / T), T, 0, stream>>>(xr2, den2, bias2, N, H2, LC2, D2);
  // h2 lives in xr2

  // ---- pool + head ----
  fill_kernel<<<1, T, 0, stream>>>(pool, (size_t)D2, 0.f);
  colsum_kernel<<<256, 512, 0, stream>>>(xr2, pool, N, D2);
  head_kernel<<<1, 128, 0, stream>>>(pool, Wlin, blin, out, D2, 128, 1.0f / (float)N);
}
